// SelfAttention_23845658428141
// MI455X (gfx1250) — compile-verified, matrix operand order checked by reading, not run on hardware
//
#include <hip/hip_runtime.h>
#include <hip/hip_bf16.h>

// MI455X (gfx1250) flash-attention pipeline, bf16 WMMA w/ fp32 accumulate.
// ~41 GFLOP, ~34 MB true HBM traffic -> matrix-unit bound; never materialize
// the 268 MB score matrix (exceeds 192 MB L2). Softmax row-sums run on the
// matrix units (P @ ones); row max uses a step-major 4-step ds_swizzle tree.
// Fragment working set bounded to 2 at a time + launch_bounds(128,1) so the
// allocator never spills the hot loop to scratch.

typedef __attribute__((ext_vector_type(16))) __bf16 v16bf;
typedef __attribute__((ext_vector_type(8)))  __bf16 v8bf;
typedef __attribute__((ext_vector_type(8)))  float  v8f;

#define BATCH 4
#define NPIX  4096   // 64*64
#define CCH   256
#define DDIM  32

union FragU { v16bf v; v8bf h[2]; };

// 16x32 bf16 fragment (A-layout; B uses same pattern with M<->N) from a
// K-major matrix with leading dimension `ld` (elements). Two 128-bit loads.
//   lanes 0-15 : row = lane,    K = 0..7 (elems0-7) and 16..23 (elems8-15)
//   lanes16-31 : row = lane-16, K = 8..15           and 24..31
__device__ __forceinline__ v16bf load_frag(const __bf16* base, int ld, int lane) {
    int r = lane & 15;
    int h = (lane >> 4) & 1;
    const __bf16* p = base + (size_t)r * ld + h * 8;
    FragU u;
    u.h[0] = *reinterpret_cast<const v8bf*>(p);
    u.h[1] = *reinterpret_cast<const v8bf*>(p + 16);
    return u.v;
}

// Raw xor-swizzle (group-of-32 pattern; and=0x1f, or=0, xor=IMM>>10).
template <int IMM>
__device__ __forceinline__ float swz(float v) {
    return __int_as_float(__builtin_amdgcn_ds_swizzle(__float_as_int(v), IMM));
}

__global__ void cvt_f32_to_bf16(const float* __restrict__ in, __bf16* __restrict__ out, int n) {
    int i = blockIdx.x * blockDim.x + threadIdx.x;
    int stride = gridDim.x * blockDim.x;
    for (; i < n; i += stride) out[i] = (__bf16)in[i];
}

// out[c][r] = in[r][c]  (out ld = rows) -> K-major bf16 weights
__global__ void transpose_to_bf16(const float* __restrict__ in, __bf16* __restrict__ out,
                                  int rows, int cols) {
    int i = blockIdx.x * blockDim.x + threadIdx.x;
    if (i >= rows * cols) return;
    int r = i / cols, c = i % cols;
    out[(size_t)c * rows + r] = (__bf16)in[i];
}

// C = A[M,K] * B with B transposed/K-major (BT[Ncols,K]). 4 waves per block;
// wave w computes M-tile (blockIdx.x*4+w) x NT consecutive N-tiles, reusing
// the A fragment NT times per K-step. TSTORE: store transposed per batch.
template <int NT, bool TSTORE>
__global__ void __launch_bounds__(128, 1)
proj_gemm(const __bf16* __restrict__ A, const __bf16* __restrict__ BT,
          __bf16* __restrict__ out, int M, int Ncols, int K, int batchN) {
    int lane = threadIdx.x & 31;
    int wave = threadIdx.x >> 5;
    int M0 = (blockIdx.x * 4 + wave) * 16;
    int N0 = blockIdx.y * 16 * NT;
    if (M0 >= M) return;

    v8f acc[NT];
#pragma unroll
    for (int t = 0; t < NT; t++) acc[t] = v8f{};

    for (int k = 0; k < K; k += 32) {
        v16bf a = load_frag(A + (size_t)M0 * K + k, K, lane);
        v16bf bfr[NT];
#pragma unroll
        for (int t = 0; t < NT; t++)
            bfr[t] = load_frag(BT + (size_t)(N0 + t * 16) * K + k, K, lane);
#pragma unroll
        for (int t = 0; t < NT; t++)
            acc[t] = __builtin_amdgcn_wmma_f32_16x16x32_bf16(false, a, false, bfr[t],
                                                             (short)0, acc[t], false, false);
    }

    int rb = ((lane >> 4) & 1) * 8;   // rows M0+rb .. M0+rb+7 in this half-wave
#pragma unroll
    for (int t = 0; t < NT; t++) {
        int col = N0 + t * 16 + (lane & 15);
        if (TSTORE) {
            int bidx = M0 / batchN;
            int m    = (M0 % batchN) + rb;
            v8bf pk;
#pragma unroll
            for (int i = 0; i < 8; i++) pk[i] = (__bf16)acc[t][i];
            *reinterpret_cast<v8bf*>(out + ((size_t)bidx * Ncols + col) * batchN + m) = pk;
        } else {
#pragma unroll
            for (int i = 0; i < 8; i++)
                out[(size_t)(M0 + rb + i) * Ncols + col] = (__bf16)acc[t][i];
        }
    }
}

// Flash attention + epilogue. Block = 4 waves = one 16-query tile of one batch.
// Wave w owns output channels [64w, 64w+64). gb/fb: [B,N,D] bf16 row-major,
// hbT: [B,C,N] bf16 (K-major over keys). 64 keys per iteration.
__global__ void __launch_bounds__(128, 1)
flash_attn(const __bf16* __restrict__ gb, const __bf16* __restrict__ fb,
           const __bf16* __restrict__ hbT, const float* __restrict__ x,
           const float* __restrict__ gammap, float* __restrict__ out) {
    __shared__ alignas(32) __bf16 Pl[4][16 * 64];  // per-wave P staging tile (2 KB)

    int lane = threadIdx.x & 31;
    int wave = threadIdx.x >> 5;
    int M0 = blockIdx.x * 16;
    int b  = blockIdx.y;
    int C0 = wave * 64;

    // Query fragment: 16 rows x K=32 (= full D) -- loaded once, reused all blocks.
    v16bf ga = load_frag(gb + ((size_t)b * NPIX + M0) * DDIM, DDIM, lane);

    // All-ones B fragment: row-sum of P via WMMA (softmax denominator).
    v16bf ones;
#pragma unroll
    for (int i = 0; i < 16; i++) ones[i] = (__bf16)1.0f;

    v8f acc[4];
#pragma unroll
    for (int t = 0; t < 4; t++) acc[t] = v8f{};
    v8f accl = {};                 // running row-sum accumulator (all 16 cols equal)
    float mrow[8];
#pragma unroll
    for (int i = 0; i < 8; i++) mrow[i] = -3.0e38f;

    const int colk = lane & 15;
    const int rb   = ((lane >> 4) & 1) * 8;
    __bf16* pw = Pl[wave];

    for (int kb = 0; kb < NPIX; kb += 64) {
        if (kb + 64 < NPIX)
            __builtin_prefetch(fb + ((size_t)b * NPIX + kb + 64) * DDIM + lane * 8, 0, 1);

        const __bf16* fbb = fb + ((size_t)b * NPIX + kb) * DDIM;
        const __bf16* hbase = hbT + ((size_t)b * CCH + C0) * NPIX + kb;

        // S tile: 16 queries x 64 keys, f fragments in pairs (2 live at once).
        v8f s[4];
        {
            v8f z = {};
            v16bf f0 = load_frag(fbb,             DDIM, lane);
            v16bf f1 = load_frag(fbb + 16 * DDIM, DDIM, lane);
            s[0] = __builtin_amdgcn_wmma_f32_16x16x32_bf16(false, ga, false, f0, (short)0, z, false, false);
            s[1] = __builtin_amdgcn_wmma_f32_16x16x32_bf16(false, ga, false, f1, (short)0, z, false, false);
            f0 = load_frag(fbb + 32 * DDIM, DDIM, lane);
            f1 = load_frag(fbb + 48 * DDIM, DDIM, lane);
            s[2] = __builtin_amdgcn_wmma_f32_16x16x32_bf16(false, ga, false, f0, (short)0, z, false, false);
            s[3] = __builtin_amdgcn_wmma_f32_16x16x32_bf16(false, ga, false, f1, (short)0, z, false, false);
        }

        // Hoist first H pair -- overlaps the global loads with softmax VALU.
        v16bf h0 = load_frag(hbase,                  NPIX, lane);
        v16bf h1 = load_frag(hbase + (size_t)16 * NPIX, NPIX, lane);

        // Row max over 64 keys: lane-local max-of-4, then 4 step-major
        // ds_swizzle stages with all 8 rows in flight per stage.
        float vmax[8], tmp[8];
#pragma unroll
        for (int i = 0; i < 8; i++)
            vmax[i] = fmaxf(fmaxf(s[0][i], s[1][i]), fmaxf(s[2][i], s[3][i]));
#pragma unroll
        for (int i = 0; i < 8; i++) tmp[i] = swz<0x041F>(vmax[i]);
#pragma unroll
        for (int i = 0; i < 8; i++) vmax[i] = fmaxf(vmax[i], tmp[i]);
#pragma unroll
        for (int i = 0; i < 8; i++) tmp[i] = swz<0x081F>(vmax[i]);
#pragma unroll
        for (int i = 0; i < 8; i++) vmax[i] = fmaxf(vmax[i], tmp[i]);
#pragma unroll
        for (int i = 0; i < 8; i++) tmp[i] = swz<0x101F>(vmax[i]);
#pragma unroll
        for (int i = 0; i < 8; i++) vmax[i] = fmaxf(vmax[i], tmp[i]);
#pragma unroll
        for (int i = 0; i < 8; i++) tmp[i] = swz<0x201F>(vmax[i]);
#pragma unroll
        for (int i = 0; i < 8; i++) vmax[i] = fmaxf(vmax[i], tmp[i]);

        // Online softmax: rescale accumulators, exponentiate scores.
#pragma unroll
        for (int i = 0; i < 8; i++) {
            float mnew  = fmaxf(mrow[i], vmax[i]);
            float scale = __expf(mrow[i] - mnew);
            mrow[i] = mnew;
#pragma unroll
            for (int c = 0; c < 4; c++) s[c][i] = __expf(s[c][i] - mnew);
#pragma unroll
            for (int t = 0; t < 4; t++) acc[t][i] *= scale;
            accl[i] *= scale;
        }

        // C-layout -> A-layout via per-wave LDS tile (row-major 16x64 bf16).
#pragma unroll
        for (int i = 0; i < 8; i++) {
#pragma unroll
            for (int c = 0; c < 4; c++)
                pw[(rb + i) * 64 + colk + c * 16] = (__bf16)s[c][i];
        }
        __builtin_amdgcn_wave_barrier();            // order stores before reads
        v16bf pa0 = load_frag(pw,      64, lane);   // keys kb+0  .. kb+31
        v16bf pa1 = load_frag(pw + 32, 64, lane);   // keys kb+32 .. kb+63
        __builtin_amdgcn_wave_barrier();            // order reads before next overwrite

        // Row sums: accl += P * ones  (2 WMMAs replace 64 shuffle ops).
        accl = __builtin_amdgcn_wmma_f32_16x16x32_bf16(false, pa0, false, ones,
                                                       (short)0, accl, false, false);
        accl = __builtin_amdgcn_wmma_f32_16x16x32_bf16(false, pa1, false, ones,
                                                       (short)0, accl, false, false);

        // acc += P(16x64) * H(64 keys x 64 ch): 8 WMMAs, 2 H fragments live.
        acc[0] = __builtin_amdgcn_wmma_f32_16x16x32_bf16(false, pa0, false, h0, (short)0, acc[0], false, false);
        acc[1] = __builtin_amdgcn_wmma_f32_16x16x32_bf16(false, pa0, false, h1, (short)0, acc[1], false, false);
        h0 = load_frag(hbase + (size_t)32 * NPIX, NPIX, lane);
        h1 = load_frag(hbase + (size_t)48 * NPIX, NPIX, lane);
        acc[2] = __builtin_amdgcn_wmma_f32_16x16x32_bf16(false, pa0, false, h0, (short)0, acc[2], false, false);
        acc[3] = __builtin_amdgcn_wmma_f32_16x16x32_bf16(false, pa0, false, h1, (short)0, acc[3], false, false);
        h0 = load_frag(hbase + 32,                      NPIX, lane);
        h1 = load_frag(hbase + (size_t)16 * NPIX + 32,  NPIX, lane);
        acc[0] = __builtin_amdgcn_wmma_f32_16x16x32_bf16(false, pa1, false, h0, (short)0, acc[0], false, false);
        acc[1] = __builtin_amdgcn_wmma_f32_16x16x32_bf16(false, pa1, false, h1, (short)0, acc[1], false, false);
        h0 = load_frag(hbase + (size_t)32 * NPIX + 32,  NPIX, lane);
        h1 = load_frag(hbase + (size_t)48 * NPIX + 32,  NPIX, lane);
        acc[2] = __builtin_amdgcn_wmma_f32_16x16x32_bf16(false, pa1, false, h0, (short)0, acc[2], false, false);
        acc[3] = __builtin_amdgcn_wmma_f32_16x16x32_bf16(false, pa1, false, h1, (short)0, acc[3], false, false);
    }

    // Epilogue: out = gamma * (acc / rowsum) + x   (fp32)
    float gamma = *gammap;
#pragma unroll
    for (int t = 0; t < 4; t++) {
        int col = C0 + t * 16 + colk;
#pragma unroll
        for (int i = 0; i < 8; i++) {
            size_t idx = ((size_t)b * NPIX + M0 + rb + i) * CCH + col;
            out[idx] = gamma * (acc[t][i] / accl[i]) + x[idx];
        }
    }
}

extern "C" void kernel_launch(void* const* d_in, const int* in_sizes, int n_in,
                              void* d_out, int out_size, void* d_ws, size_t ws_size,
                              hipStream_t stream) {
    const float* x     = (const float*)d_in[0];
    const float* Wf    = (const float*)d_in[1];
    const float* Wg    = (const float*)d_in[2];
    const float* Wh    = (const float*)d_in[3];
    const float* gamma = (const float*)d_in[4];
    float* out = (float*)d_out;

    char* ws = (char*)d_ws;
    // 256B-aligned workspace carve-up (total ~19.1 MB)
    __bf16* xb  = (__bf16*)(ws + 0);          // [B*N, C]      8 MB
    __bf16* hbT = (__bf16*)(ws + 8388608);    // [B, C, N]     8 MB
    __bf16* fb  = (__bf16*)(ws + 16777216);   // [B*N, D]      1 MB
    __bf16* gb  = (__bf16*)(ws + 17825792);   // [B*N, D]      1 MB
    __bf16* WfT = (__bf16*)(ws + 18874368);   // [D, C]
    __bf16* WgT = (__bf16*)(ws + 18890752);   // [D, C]
    __bf16* WhT = (__bf16*)(ws + 18907136);   // [C, C]

    const int M = BATCH * NPIX;               // 16384

    // Stage 1: bf16 conversion + K-major weight transposes
    cvt_f32_to_bf16<<<1024, 256, 0, stream>>>(x, xb, M * CCH);
    transpose_to_bf16<<<(CCH * DDIM + 255) / 256, 256, 0, stream>>>(Wf, WfT, CCH, DDIM);
    transpose_to_bf16<<<(CCH * DDIM + 255) / 256, 256, 0, stream>>>(Wg, WgT, CCH, DDIM);
    transpose_to_bf16<<<(CCH * CCH + 255) / 256, 256, 0, stream>>>(Wh, WhT, CCH, CCH);

    // Stage 2: projections (WMMA GEMMs). f,g row-major; h stored transposed,
    // with NT N-tiles per wave to reuse the A fragment.
    proj_gemm<2, false><<<dim3(M / 64, 1), 128, 0, stream>>>(xb, WfT, fb, M, DDIM, CCH, NPIX);
    proj_gemm<2, false><<<dim3(M / 64, 1), 128, 0, stream>>>(xb, WgT, gb, M, DDIM, CCH, NPIX);
    proj_gemm<4, true ><<<dim3(M / 64, 4), 128, 0, stream>>>(xb, WhT, hbT, M, CCH, CCH, NPIX);

    // Stage 3: flash attention + residual epilogue
    flash_attn<<<dim3(NPIX / 16, BATCH), 128, 0, stream>>>(gb, fb, hbT, x, gamma, out);
}
